// DGIWithGIN_66340064854116
// MI455X (gfx1250) — matrix operand order repaired and checked by reading, not compiled, not run on hardware
//
#include <hip/hip_runtime.h>
#include <hip/hip_bf16.h>

typedef __bf16 bf16_t;
typedef __attribute__((ext_vector_type(16))) __bf16 v16bf;
typedef __attribute__((ext_vector_type(8)))  __bf16 v8bf;
typedef __attribute__((ext_vector_type(8)))  float  v8f;

// ---------------------------------------------------------------------------
// Elementwise / copy helpers
// ---------------------------------------------------------------------------
__global__ __launch_bounds__(256) void dgi_copy_f32(float* __restrict__ dst,
                                                    const float* __restrict__ src, int n) {
  int i = blockIdx.x * 256 + threadIdx.x;
  if (i < n) dst[i] = src[i];
}

__global__ __launch_bounds__(256) void dgi_zero_f32(float* __restrict__ dst, int n) {
  int i = blockIdx.x * blockDim.x + threadIdx.x;
  if (i < n) dst[i] = 0.0f;
}

__global__ __launch_bounds__(256) void dgi_gather_rows(float* __restrict__ dst,
                                                       const float* __restrict__ src,
                                                       const int* __restrict__ perm,
                                                       int K, int n) {
  int i = blockIdx.x * 256 + threadIdx.x;
  if (i >= n) return;
  int r = i / K, f = i - r * K;
  dst[i] = src[(size_t)perm[r] * K + f];
}

// ---------------------------------------------------------------------------
// Edge scatter-add: agg[dst] += x[src]  (agg pre-initialized with x -> GIN h)
// One block per edge, threads stride features; native f32 atomics (L2 resident)
// ---------------------------------------------------------------------------
__global__ __launch_bounds__(128) void dgi_scatter_add(const int* __restrict__ src,
                                                       const int* __restrict__ dst,
                                                       const float* __restrict__ x,
                                                       float* __restrict__ agg, int K) {
  int e = blockIdx.x;
  int s = src[e], d = dst[e];
  const float* xs = x + (size_t)s * K;
  float* ad = agg + (size_t)d * K;
  for (int f = threadIdx.x; f < K; f += 128)
    unsafeAtomicAdd(&ad[f], xs[f]);
}

// ---------------------------------------------------------------------------
// fp32 -> bf16 with zero padding of K to Kpad (multiple of 32 for WMMA)
// ---------------------------------------------------------------------------
__global__ __launch_bounds__(256) void dgi_to_bf16_pad(const float* __restrict__ in,
                                                       bf16_t* __restrict__ out,
                                                       int K, int Kpad, int n) {
  int i = blockIdx.x * 256 + threadIdx.x;
  if (i >= n) return;
  int r = i / Kpad, k = i - r * Kpad;
  out[i] = (k < K) ? (bf16_t)in[(size_t)r * K + k] : (bf16_t)0.0f;
}

// Weight [K,128] fp32 -> transposed bf16 [128, Kpad] (so B frags load contiguously)
__global__ __launch_bounds__(256) void dgi_conv_weight(const float* __restrict__ W,
                                                       bf16_t* __restrict__ Wt,
                                                       int K, int Kpad) {
  int i = blockIdx.x * 256 + threadIdx.x;
  if (i >= 128 * Kpad) return;
  int nn = i / Kpad, k = i - nn * Kpad;
  Wt[i] = (k < K) ? (bf16_t)W[k * 128 + nn] : (bf16_t)0.0f;
}

// ---------------------------------------------------------------------------
// WMMA GEMM: C[M,128] = A[M,K](bf16) @ Bt[128,K](bf16, pre-transposed), f32 acc
// Block: 256 thr = 8 waves; tile 128x128; wave tile 32x64 (2x4 WMMA frags)
// ---------------------------------------------------------------------------
__global__ __launch_bounds__(256) void dgi_gemm_wmma(const bf16_t* __restrict__ A,
                                                     const bf16_t* __restrict__ Bt,
                                                     float* __restrict__ C, int K) {
  const int lane = threadIdx.x & 31;
  const int wave = threadIdx.x >> 5;
  const int wrow = blockIdx.x * 128 + (wave & 3) * 32;
  const int wcol = (wave >> 2) * 64;
  const int lrow = lane & 15;
  const int lhi  = lane >> 4;

  v8f acc[2][4];
#pragma unroll
  for (int r = 0; r < 2; ++r)
#pragma unroll
    for (int c = 0; c < 4; ++c)
#pragma unroll
      for (int i = 0; i < 8; ++i) acc[r][c][i] = 0.0f;

  for (int k0 = 0; k0 < K; k0 += 32) {
    // 16-bit A layout: lanes 0-15 -> K {0..7,16..23}; lanes 16-31 -> K {8..15,24..31}
    v16bf afrag[2], bfrag[4];
#pragma unroll
    for (int r = 0; r < 2; ++r) {
      const bf16_t* ap = A + (size_t)(wrow + r * 16 + lrow) * K + k0 + lhi * 8;
      v8bf lo = *(const v8bf*)(ap);
      v8bf hi = *(const v8bf*)(ap + 16);
#pragma unroll
      for (int i = 0; i < 8; ++i) { afrag[r][i] = lo[i]; afrag[r][8 + i] = hi[i]; }
    }
    // 16-bit B layout: lanes 0-15 hold K {0..7,16..23} of column lrow; lanes 16-31 hold K {8..15,24..31}
#pragma unroll
    for (int c = 0; c < 4; ++c) {
      const bf16_t* bp = Bt + (size_t)(wcol + c * 16 + lrow) * K + k0 + lhi * 8;
      v8bf lo = *(const v8bf*)(bp);
      v8bf hi = *(const v8bf*)(bp + 16);
#pragma unroll
      for (int i = 0; i < 8; ++i) { bfrag[c][i] = lo[i]; bfrag[c][8 + i] = hi[i]; }
    }
#pragma unroll
    for (int r = 0; r < 2; ++r)
#pragma unroll
      for (int c = 0; c < 4; ++c)
        acc[r][c] = __builtin_amdgcn_wmma_f32_16x16x32_bf16(
            false, afrag[r], false, bfrag[c], (short)0, acc[r][c], false, false);
  }

#pragma unroll
  for (int r = 0; r < 2; ++r)
#pragma unroll
    for (int c = 0; c < 4; ++c) {
      int col = wcol + c * 16 + lrow;
#pragma unroll
      for (int v = 0; v < 8; ++v) {
        int row = wrow + r * 16 + v + lhi * 8;
        C[(size_t)row * 128 + col] = acc[r][c][v];
      }
    }
}

// ---------------------------------------------------------------------------
// BatchNorm (training-mode batch stats): column sums/sumsq then apply
// ---------------------------------------------------------------------------
__global__ __launch_bounds__(128) void dgi_col_stats(const float* __restrict__ Z,
                                                     float* __restrict__ sums,
                                                     float* __restrict__ sumsq, int M) {
  int col = threadIdx.x;  // 128 columns
  float s = 0.f, q = 0.f;
  for (int r = blockIdx.x; r < M; r += gridDim.x) {
    float v = Z[(size_t)r * 128 + col];
    s += v; q += v * v;
  }
  unsafeAtomicAdd(&sums[col], s);
  unsafeAtomicAdd(&sumsq[col], q);
}

__global__ __launch_bounds__(256) void dgi_bn_apply(const float* __restrict__ Z,
                                                    const float* __restrict__ sums,
                                                    const float* __restrict__ sumsq,
                                                    const float* __restrict__ gamma,
                                                    const float* __restrict__ beta,
                                                    float* __restrict__ outF,
                                                    bf16_t* __restrict__ outB,
                                                    int M, int relu) {
  int i = blockIdx.x * 256 + threadIdx.x;
  if (i >= M * 128) return;
  int col = i & 127;
  float invM  = 1.0f / (float)M;
  float mean  = sums[col] * invM;
  float var   = sumsq[col] * invM - mean * mean;
  float scale = gamma[col] * rsqrtf(var + 1e-5f);
  float v = (Z[i] - mean) * scale + beta[col];
  if (relu) v = fmaxf(v, 0.0f);
  if (outF) outF[i] = v;
  if (outB) outB[i] = (bf16_t)v;
}

// ---------------------------------------------------------------------------
// Pooling + summary MLP + bilinear fold
// ---------------------------------------------------------------------------
__global__ __launch_bounds__(256) void dgi_pool_count(const int* __restrict__ batch,
                                                      float* __restrict__ cnt, int N) {
  int n = blockIdx.x * 256 + threadIdx.x;
  if (n < N) unsafeAtomicAdd(&cnt[batch[n]], 1.0f);
}

__global__ __launch_bounds__(256) void dgi_pool_sum(const float* __restrict__ pos,
                                                    const int* __restrict__ batch,
                                                    float* __restrict__ psum, int N) {
  int i = blockIdx.x * 256 + threadIdx.x;
  if (i >= N * 128) return;
  int n = i >> 7, h = i & 127;
  unsafeAtomicAdd(&psum[batch[n] * 128 + h], pos[i]);
}

// Per graph g: summ = psum/cnt; s = sigmoid(relu(summ@sW1+sb1)@sW2+sb2);
// WsT[g,h] = sum_k dW[h,k] * s[g,k]  (folds the bilinear form)
__global__ __launch_bounds__(128) void dgi_summary(const float* __restrict__ psum,
                                                   const float* __restrict__ pcnt,
                                                   const float* __restrict__ sW1,
                                                   const float* __restrict__ sb1,
                                                   const float* __restrict__ sW2,
                                                   const float* __restrict__ sb2,
                                                   const float* __restrict__ dW,
                                                   float* __restrict__ WsT) {
  __shared__ float sm[128], t1[128], s2[128];
  int g = blockIdx.x, h = threadIdx.x;
  float cnt = fmaxf(pcnt[g], 1.0f);
  sm[h] = psum[g * 128 + h] / cnt;
  __syncthreads();
  float a = sb1[h];
  for (int k = 0; k < 128; ++k) a += sm[k] * sW1[k * 128 + h];
  t1[h] = fmaxf(a, 0.0f);
  __syncthreads();
  float b = sb2[h];
  for (int k = 0; k < 128; ++k) b += t1[k] * sW2[k * 128 + h];
  s2[h] = 1.0f / (1.0f + __expf(-b));
  __syncthreads();
  float w = 0.0f;
  for (int k = 0; k < 128; ++k) w += dW[h * 128 + k] * s2[k];
  WsT[g * 128 + h] = w;
}

// ---------------------------------------------------------------------------
// Scores: one wave32 per node; lane reads float4 (coalesced 512B/row), xor-reduce
// ---------------------------------------------------------------------------
__global__ __launch_bounds__(256) void dgi_scores(const float* __restrict__ pos,
                                                  const float* __restrict__ neg,
                                                  const float* __restrict__ WsT,
                                                  const int* __restrict__ batch,
                                                  const float* __restrict__ db,
                                                  float* __restrict__ out, int N) {
  int wave = threadIdx.x >> 5, lane = threadIdx.x & 31;
  int n = blockIdx.x * 8 + wave;
  if (n >= N) return;
  int g = batch[n];
  float4 wv = ((const float4*)(WsT + g * 128))[lane];
  float4 pv = ((const float4*)(pos + (size_t)n * 128))[lane];
  float4 qv = ((const float4*)(neg + (size_t)n * 128))[lane];
  float sp = wv.x * pv.x + wv.y * pv.y + wv.z * pv.z + wv.w * pv.w;
  float sn = wv.x * qv.x + wv.y * qv.y + wv.z * qv.z + wv.w * qv.w;
#pragma unroll
  for (int off = 16; off > 0; off >>= 1) {
    sp += __shfl_xor(sp, off, 32);
    sn += __shfl_xor(sn, off, 32);
  }
  if (lane == 0) {
    float bias = db[0];
    out[n] = sp + bias;
    out[N + n] = sn + bias;
  }
}

// ---------------------------------------------------------------------------
// Host orchestration
// ---------------------------------------------------------------------------
extern "C" void kernel_launch(void* const* d_in, const int* in_sizes, int n_in,
                              void* d_out, int out_size, void* d_ws, size_t ws_size,
                              hipStream_t stream) {
  const float* x     = (const float*)d_in[0];
  const int*   edge  = (const int*)d_in[1];
  const int*   batch = (const int*)d_in[2];
  const int*   perm  = (const int*)d_in[3];

  const int N   = in_sizes[2];
  const int E   = in_sizes[1] / 2;
  const int FIN = in_sizes[0] / N;   // 165
  const int H = 128, G = 64;
  const int K0PAD = ((FIN + 31) / 32) * 32;  // 192

  const int* srcI = edge;
  const int* dstI = edge + E;

  const float *W1[3], *g1[3], *be1[3], *W2[3], *g2[3], *be2[3];
  for (int l = 0; l < 3; ++l) {
    int base = 4 + l * 8;                 // {W1,b1,g1,be1,W2,b2,g2,be2}; biases cancel under BN
    W1[l]  = (const float*)d_in[base + 0];
    g1[l]  = (const float*)d_in[base + 2];
    be1[l] = (const float*)d_in[base + 3];
    W2[l]  = (const float*)d_in[base + 4];
    g2[l]  = (const float*)d_in[base + 6];
    be2[l] = (const float*)d_in[base + 7];
  }
  const float* sW1 = (const float*)d_in[28];
  const float* sb1 = (const float*)d_in[29];
  const float* sW2 = (const float*)d_in[30];
  const float* sb2 = (const float*)d_in[31];
  const float* dW  = (const float*)d_in[32];
  const float* db  = (const float*)d_in[33];
  (void)n_in; (void)out_size; (void)ws_size;

  // Workspace carve-out (all tensors L2-sized; MI455X L2 = 192MB keeps actives hot)
  char* ws = (char*)d_ws;
  size_t off = 0;
  auto alloc = [&](size_t bytes) -> void* {
    off = (off + 255) & ~(size_t)255;
    void* p = ws + off;
    off += bytes;
    return p;
  };
  bf16_t* Wt1[3]; bf16_t* Wt2[3];
  Wt1[0] = (bf16_t*)alloc((size_t)H * K0PAD * sizeof(bf16_t));
  Wt2[0] = (bf16_t*)alloc((size_t)H * H * sizeof(bf16_t));
  for (int l = 1; l < 3; ++l) {
    Wt1[l] = (bf16_t*)alloc((size_t)H * H * sizeof(bf16_t));
    Wt2[l] = (bf16_t*)alloc((size_t)H * H * sizeof(bf16_t));
  }
  bf16_t* Abf  = (bf16_t*)alloc((size_t)N * K0PAD * sizeof(bf16_t));
  float*  Z    = (float*)alloc((size_t)N * H * sizeof(float));
  float*  agg  = (float*)alloc((size_t)N * FIN * sizeof(float));
  float*  actA = (float*)alloc((size_t)N * FIN * sizeof(float));
  float*  actB = (float*)alloc((size_t)N * FIN * sizeof(float));
  float*  posb = (float*)alloc((size_t)N * H * sizeof(float));
  float*  stats = (float*)alloc(256 * sizeof(float));   // sums[128] | sumsq[128]
  float*  psum  = (float*)alloc((size_t)G * H * sizeof(float));
  float*  pcnt  = (float*)alloc((size_t)G * sizeof(float));
  float*  WsT   = (float*)alloc((size_t)G * H * sizeof(float));

  // Weights -> transposed bf16 (padded K)
  {
    int n0 = H * K0PAD, n1 = H * H;
    dgi_conv_weight<<<(n0 + 255) / 256, 256, 0, stream>>>(W1[0], Wt1[0], FIN, K0PAD);
    dgi_conv_weight<<<(n1 + 255) / 256, 256, 0, stream>>>(W2[0], Wt2[0], H, H);
    for (int l = 1; l < 3; ++l) {
      dgi_conv_weight<<<(n1 + 255) / 256, 256, 0, stream>>>(W1[l], Wt1[l], H, H);
      dgi_conv_weight<<<(n1 + 255) / 256, 256, 0, stream>>>(W2[l], Wt2[l], H, H);
    }
  }

  auto run_layer = [&](const float* cur, int Kin, int Kpad, int layer,
                       float* outF, int relu_last) {
    int nel = N * Kin;
    dgi_copy_f32<<<(nel + 255) / 256, 256, 0, stream>>>(agg, cur, nel);           // h = x + ...
    dgi_scatter_add<<<E, 128, 0, stream>>>(srcI, dstI, cur, agg, Kin);            // ... + sum_j x_j
    int npad = N * Kpad;
    dgi_to_bf16_pad<<<(npad + 255) / 256, 256, 0, stream>>>(agg, Abf, Kin, Kpad, npad);
    dgi_gemm_wmma<<<N / 128, 256, 0, stream>>>(Abf, Wt1[layer], Z, Kpad);
    dgi_zero_f32<<<1, 256, 0, stream>>>(stats, 256);
    dgi_col_stats<<<256, 128, 0, stream>>>(Z, stats, stats + 128, N);
    dgi_bn_apply<<<(N * H + 255) / 256, 256, 0, stream>>>(
        Z, stats, stats + 128, g1[layer], be1[layer], nullptr, Abf, N, 1);
    dgi_gemm_wmma<<<N / 128, 256, 0, stream>>>(Abf, Wt2[layer], Z, H);
    dgi_zero_f32<<<1, 256, 0, stream>>>(stats, 256);
    dgi_col_stats<<<256, 128, 0, stream>>>(Z, stats, stats + 128, N);
    dgi_bn_apply<<<(N * H + 255) / 256, 256, 0, stream>>>(
        Z, stats, stats + 128, g2[layer], be2[layer], outF, nullptr, N, relu_last);
  };

  // --- positive encode: x -> posb ---
  run_layer(x,    FIN, K0PAD, 0, actA, 1);
  run_layer(actA, H,   H,     1, actB, 1);
  run_layer(actB, H,   H,     2, posb, 0);

  // --- mean-pool + summary MLP + bilinear fold ---
  dgi_zero_f32<<<(G * H + 255) / 256, 256, 0, stream>>>(psum, G * H);
  dgi_zero_f32<<<1, 64, 0, stream>>>(pcnt, G);
  dgi_pool_count<<<(N + 255) / 256, 256, 0, stream>>>(batch, pcnt, N);
  dgi_pool_sum<<<(N * H + 255) / 256, 256, 0, stream>>>(posb, batch, psum, N);
  dgi_summary<<<G, 128, 0, stream>>>(psum, pcnt, sW1, sb1, sW2, sb2, dW, WsT);

  // --- negative encode: x[perm] -> actB ---
  {
    int nel0 = N * FIN;
    dgi_gather_rows<<<(nel0 + 255) / 256, 256, 0, stream>>>(actA, x, perm, FIN, nel0);
  }
  run_layer(actA, FIN, K0PAD, 0, actB, 1);
  run_layer(actB, H,   H,     1, actA, 1);
  run_layer(actA, H,   H,     2, actB, 0);   // neg final

  // --- scores ---
  dgi_scores<<<(N + 7) / 8, 256, 0, stream>>>(posb, actB, WsT, batch, db,
                                              (float*)d_out, N);
}
